// GraphConv_22488448761919
// MI455X (gfx1250) — matrix operand order repaired and checked by reading, not compiled, not run on hardware
//
#include <hip/hip_runtime.h>

// Problem sizes (fixed by the reference)
#define BB   8
#define FIN  512
#define FOUT 512
#define NN   4096

typedef __bf16 bf16_t;
typedef __attribute__((ext_vector_type(16))) bf16_t v16bf;
typedef __attribute__((ext_vector_type(8)))  float  v8f;
typedef __attribute__((ext_vector_type(4)))  float  v4f;

// float -> bf16, round-to-nearest-even (used only in preprocessing kernels)
static __device__ __forceinline__ bf16_t f2bf(float f) {
  unsigned int u = __builtin_bit_cast(unsigned int, f);
  u += 0x7FFFu + ((u >> 16) & 1u);
  unsigned short s = (unsigned short)(u >> 16);
  return __builtin_bit_cast(bf16_t, s);
}

// K-swizzle within a 32-wide block so each half-wave's A fragment is one
// contiguous 16-element run: [0..7,16..23 | 8..15,24..31]
static __device__ __forceinline__ int swz(int q) {
  return (q & 7) | ((q & 8) << 1) | ((q & 16) >> 1);
}

// ---------------------------------------------------------------------------
// inv_cs[j] = 1 / colsum[j] (1.0 where colsum==0), colsum[j] = sum_k adj[k][j]
// ---------------------------------------------------------------------------
__global__ void __launch_bounds__(256) colsum_inv_kernel(const float* __restrict__ adj,
                                                         float* __restrict__ inv_cs) {
  const int j = blockIdx.x * blockDim.x + threadIdx.x;
  if (j >= NN) return;
  float s = 0.0f;
  for (int k = 0; k < NN; ++k) s += adj[(size_t)k * NN + j];  // coalesced across threads
  inv_cs[j] = (s == 0.0f) ? 1.0f : (1.0f / s);
}

// ---------------------------------------------------------------------------
// adj (f32, row-major) -> adj_bf (bf16, same layout). Vectorized x4.
// ---------------------------------------------------------------------------
__global__ void __launch_bounds__(256) cvt_adj_kernel(const float* __restrict__ adj,
                                                      bf16_t* __restrict__ adj_bf) {
  const size_t i4 = ((size_t)blockIdx.x * blockDim.x + threadIdx.x) * 4;
  v4f v = *(const v4f*)(adj + i4);
  bf16_t* o = adj_bf + i4;
  o[0] = f2bf(v.x); o[1] = f2bf(v.y); o[2] = f2bf(v.z); o[3] = f2bf(v.w);
}

// ---------------------------------------------------------------------------
// weight (f32 [FOUT][FIN]) -> wsw (bf16 [FOUT][FIN], plain row-major)
// ---------------------------------------------------------------------------
__global__ void __launch_bounds__(256) cvt_weight_kernel(const float* __restrict__ weight,
                                                         bf16_t* __restrict__ wsw) {
  const int idx = blockIdx.x * blockDim.x + threadIdx.x;  // o*FIN + i
  const int i   = idx & (FIN - 1);
  const int o   = idx >> 9;                               // FIN == 512
  wsw[(size_t)o * FIN + (i & ~31) + (i & 31)] = f2bf(weight[idx]);
}

// ---------------------------------------------------------------------------
// input [B][FIN][NN] f32 -> Tin [B][NN][FIN] bf16, LDS-tiled 32x32 transpose
// ---------------------------------------------------------------------------
__global__ void __launch_bounds__(256) transpose_in_kernel(const float* __restrict__ input,
                                                           bf16_t* __restrict__ Tin) {
  __shared__ float tile[32][33];                          // +1 pitch: conflict-free
  const int bid = blockIdx.x;
  const int b   = bid / ((FIN / 32) * (NN / 32));
  const int rem = bid % ((FIN / 32) * (NN / 32));
  const int i0  = (rem / (NN / 32)) * 32;
  const int n0  = (rem % (NN / 32)) * 32;

  #pragma unroll
  for (int e = 0; e < 4; ++e) {
    const int idx = threadIdx.x + e * 256;
    const int r = idx >> 5, c = idx & 31;                 // r: i, c: n (coalesced read)
    tile[r][c] = input[((size_t)b * FIN + i0 + r) * NN + n0 + c];
  }
  __syncthreads();
  #pragma unroll
  for (int e = 0; e < 4; ++e) {
    const int idx = threadIdx.x + e * 256;
    const int r = idx >> 5, c = idx & 31;                 // r: n, c: i (coalesced write)
    Tin[((size_t)b * NN + n0 + r) * FIN + i0 + c] = f2bf(tile[c][r]);
  }
}

// ---------------------------------------------------------------------------
// GEMM1: supp[b,o,n] = sum_i weight[o,i]*input[b,i,n], stored bf16 in plain order
// along n. Each wave: 32(M) x 64(N) tile, 8 WMMA/K-step, double-buffered,
// last K-step peeled (no in-loop tail select).
// ---------------------------------------------------------------------------
__global__ void __launch_bounds__(256) gemm1_kernel(const bf16_t* __restrict__ Tin,
                                                    const bf16_t* __restrict__ wsw,
                                                    bf16_t* __restrict__ supp) {
  const int lane = threadIdx.x & 31;
  const int wid  = threadIdx.x >> 5;
  const int w    = blockIdx.x * 8 + wid;
  const int b    = w / (16 * 64);                         // 16 M-tiles, 64 N-strips
  const int rem  = w % (16 * 64);
  const int o0   = (rem / 64) * 32;
  const int n0   = (rem % 64) * 64;

  const int ln = lane & 15;
  const int h  = lane >> 4;
  const int kh = 16 * h;

  v8f acc[2][4];
  #pragma unroll
  for (int u = 0; u < 2; ++u)
    #pragma unroll
    for (int t = 0; t < 4; ++t) acc[u][t] = v8f{};

  const bf16_t* pa[2];
  #pragma unroll
  for (int u = 0; u < 2; ++u)
    pa[u] = wsw + (size_t)(o0 + u * 16 + ln) * FIN + kh;
  const bf16_t* pb[4];
  #pragma unroll
  for (int t = 0; t < 4; ++t)
    pb[t] = Tin + ((size_t)b * NN + n0 + t * 16 + ln) * FIN + kh;

  v16bf ac[2], bc[4];
  #pragma unroll
  for (int u = 0; u < 2; ++u) ac[u] = *(const v16bf*)pa[u];
  #pragma unroll
  for (int t = 0; t < 4; ++t) bc[t] = *(const v16bf*)pb[t];

  #pragma unroll 2
  for (int it = 0; it < FIN / 32 - 1; ++it) {
    v16bf an[2], bn[4];
    #pragma unroll
    for (int u = 0; u < 2; ++u) { pa[u] += 32; an[u] = *(const v16bf*)pa[u]; }
    #pragma unroll
    for (int t = 0; t < 4; ++t) { pb[t] += 32; bn[t] = *(const v16bf*)pb[t]; }

    #pragma unroll
    for (int t = 0; t < 4; ++t)
      #pragma unroll
      for (int u = 0; u < 2; ++u)
        acc[u][t] = __builtin_amdgcn_wmma_f32_16x16x32_bf16(false, ac[u], false, bc[t],
                                                            (short)0, acc[u][t], false, false);
    #pragma unroll
    for (int u = 0; u < 2; ++u) ac[u] = an[u];
    #pragma unroll
    for (int t = 0; t < 4; ++t) bc[t] = bn[t];
  }
  // peeled final K-step
  #pragma unroll
  for (int t = 0; t < 4; ++t)
    #pragma unroll
    for (int u = 0; u < 2; ++u)
      acc[u][t] = __builtin_amdgcn_wmma_f32_16x16x32_bf16(false, ac[u], false, bc[t],
                                                          (short)0, acc[u][t], false, false);

  // Store D plain along n (GEMM2 consumes it as A)
  #pragma unroll
  for (int t = 0; t < 4; ++t) {
    const int q     = ((t & 1) * 16 + ln);                // n offset within 32-block
    const int n_idx = n0 + (t >> 1) * 32 + q;
    #pragma unroll
    for (int u = 0; u < 2; ++u)
      #pragma unroll
      for (int r = 0; r < 8; ++r) {
        const int o = o0 + u * 16 + r + 8 * h;
        supp[((size_t)b * FOUT + o) * NN + n_idx] = f2bf(acc[u][t][r]);
      }
  }
}

// ---------------------------------------------------------------------------
// GEMM2: out[b,o,m] = inv_cs[m] * sum_n supp[b,o,n]*adj[m,n] + bias[o]
// 32(M) x 64(N) per wave, K=4096, double-buffered, last K-step peeled.
// ---------------------------------------------------------------------------
__global__ void __launch_bounds__(256) gemm2_kernel(const bf16_t* __restrict__ supp,
                                                    const bf16_t* __restrict__ adj_bf,
                                                    const float* __restrict__ inv_cs,
                                                    const float* __restrict__ bias,
                                                    float* __restrict__ out) {
  const int lane = threadIdx.x & 31;
  const int wid  = threadIdx.x >> 5;
  const int w    = blockIdx.x * 8 + wid;
  const int b    = w / (16 * 64);
  const int rem  = w % (16 * 64);
  const int o0   = (rem / 64) * 32;
  const int m0   = (rem % 64) * 64;

  const int ln = lane & 15;
  const int h  = lane >> 4;
  const int kh = 16 * h;

  v8f acc[2][4];
  #pragma unroll
  for (int u = 0; u < 2; ++u)
    #pragma unroll
    for (int t = 0; t < 4; ++t) acc[u][t] = v8f{};

  const bf16_t* pa[2];
  #pragma unroll
  for (int u = 0; u < 2; ++u)
    pa[u] = supp + ((size_t)b * FOUT + o0 + u * 16 + ln) * NN + kh;
  const bf16_t* pb[4];
  #pragma unroll
  for (int t = 0; t < 4; ++t)
    pb[t] = adj_bf + (size_t)(m0 + t * 16 + ln) * NN + kh;

  v16bf ac[2], bc[4];
  #pragma unroll
  for (int u = 0; u < 2; ++u) ac[u] = *(const v16bf*)pa[u];
  #pragma unroll
  for (int t = 0; t < 4; ++t) bc[t] = *(const v16bf*)pb[t];

  #pragma unroll 2
  for (int it = 0; it < NN / 32 - 1; ++it) {
    v16bf an[2], bn[4];
    #pragma unroll
    for (int u = 0; u < 2; ++u) { pa[u] += 32; an[u] = *(const v16bf*)pa[u]; }
    #pragma unroll
    for (int t = 0; t < 4; ++t) { pb[t] += 32; bn[t] = *(const v16bf*)pb[t]; }

    // stream-ahead hints for the A rows (gfx1250 global_prefetch_b8)
    __builtin_prefetch(pa[0] + 256, 0, 3);
    __builtin_prefetch(pa[1] + 256, 0, 3);

    #pragma unroll
    for (int t = 0; t < 4; ++t)
      #pragma unroll
      for (int u = 0; u < 2; ++u)
        acc[u][t] = __builtin_amdgcn_wmma_f32_16x16x32_bf16(false, ac[u], false, bc[t],
                                                            (short)0, acc[u][t], false, false);
    #pragma unroll
    for (int u = 0; u < 2; ++u) ac[u] = an[u];
    #pragma unroll
    for (int t = 0; t < 4; ++t) bc[t] = bn[t];
  }
  // peeled final K-step
  #pragma unroll
  for (int t = 0; t < 4; ++t)
    #pragma unroll
    for (int u = 0; u < 2; ++u)
      acc[u][t] = __builtin_amdgcn_wmma_f32_16x16x32_bf16(false, ac[u], false, bc[t],
                                                          (short)0, acc[u][t], false, false);

  // Epilogue: per-column (m) normalization + per-row (o) bias, fp32 store
  #pragma unroll
  for (int t = 0; t < 4; ++t) {
    const int m  = m0 + t * 16 + ln;
    const float s = inv_cs[m];
    #pragma unroll
    for (int u = 0; u < 2; ++u)
      #pragma unroll
      for (int r = 0; r < 8; ++r) {
        const int o = o0 + u * 16 + r + 8 * h;
        out[((size_t)b * FOUT + o) * NN + m] = acc[u][t][r] * s + bias[o];
      }
  }
}

// ---------------------------------------------------------------------------
// Host-side launcher
// ---------------------------------------------------------------------------
extern "C" void kernel_launch(void* const* d_in, const int* in_sizes, int n_in,
                              void* d_out, int out_size, void* d_ws, size_t ws_size,
                              hipStream_t stream) {
  (void)in_sizes; (void)n_in; (void)out_size; (void)ws_size;

  const float* input  = (const float*)d_in[0];   // [B, FIN, N]
  const float* adj    = (const float*)d_in[1];   // [N, N]
  const float* weight = (const float*)d_in[2];   // [FOUT, FIN]
  const float* bias   = (const float*)d_in[3];   // [FOUT]
  float*       out    = (float*)d_out;           // [B, FOUT, N]

  // Workspace layout (1 MB alignment between regions):
  char* ws = (char*)d_ws;
  float*  inv_cs = (float*)(ws);                                   //  16 KB
  bf16_t* wsw    = (bf16_t*)(ws + (1u << 20));                     // 512 KB
  bf16_t* adj_bf = (bf16_t*)(ws + (2u << 20));                     //  32 MB
  bf16_t* Tin    = (bf16_t*)(ws + (34u << 20));                    //  32 MB
  bf16_t* supp   = (bf16_t*)(ws + (66u << 20));                    //  32 MB

  colsum_inv_kernel<<<NN / 256, 256, 0, stream>>>(adj, inv_cs);
  cvt_adj_kernel<<<(NN * (size_t)NN) / 4 / 256, 256, 0, stream>>>(adj, adj_bf);
  cvt_weight_kernel<<<(FOUT * FIN) / 256, 256, 0, stream>>>(weight, wsw);
  transpose_in_kernel<<<BB * (FIN / 32) * (NN / 32), 256, 0, stream>>>(input, Tin);

  // waves = B * (FOUT/32) * (N/64) = 8192 ; 8 waves/block -> 1024 blocks
  gemm1_kernel<<<(BB * 16 * 64) / 8, 256, 0, stream>>>(Tin, wsw, supp);
  gemm2_kernel<<<(BB * 16 * 64) / 8, 256, 0, stream>>>(supp, adj_bf, inv_cs, bias, out);
}
